// NTXentLoss_43344809951557
// MI455X (gfx1250) — matrix operand order repaired and checked by reading, not compiled, not run on hardware
//
#include <hip/hip_runtime.h>
#include <hip/hip_bf16.h>
#include <math.h>

// ---------------------------------------------------------------------------
// NT-Xent loss, fused. Pipeline:
//   1) normalize rows of [z_i; z_j] -> f16 reps (4 MB) + invnorm
//   2) exact f32 positives: pos[k] = <zi_k, zj_k> * invnorm_i * invnorm_j
//   3) fused GEMM(reps @ reps^T) + exp(2*sim) + row-sum via v_wmma_f32_16x16x32_f16
//      32-col B tiles staged in LDS (double-buffered, 528B-padded rows),
//      depth-2 software pipeline on fragment ds_loads, two interleaved
//      accumulator chains (no back-to-back WMMA RAW).
//   4) loss = mean_k( log(rowsum_k - e^2) - 2*pos_k )   [diag masked analytically]
// ---------------------------------------------------------------------------

typedef __attribute__((ext_vector_type(8)))  _Float16 v8h;
typedef __attribute__((ext_vector_type(16))) _Float16 v16h;
typedef __attribute__((ext_vector_type(8)))  float    v8f;

#define NROWS     4096
#define DIM       256
#define TWO_N     8192
#define TEMP_INV  2.0f                    // 1/temperature, T = 0.5
#define LOG2E_X2  2.885390081777927f      // 2 * log2(e)   -> exp(2x) = exp2(x*this)
#define E2        7.38905609893065f       // exp(2*1.0): self-similarity removed per row

#define ROW_H     264                     // LDS row pitch in halfs: 528B = 132 dwords
                                          // 132 % 64 = 4 -> 16 rows stripe all 64 banks

union Frag { v16h v; v8h h[2]; };

__device__ __forceinline__ float half16_sum(float v) {   // reduce within 16-lane group
  v += __shfl_xor(v, 8, 32);
  v += __shfl_xor(v, 4, 32);
  v += __shfl_xor(v, 2, 32);
  v += __shfl_xor(v, 1, 32);
  return v;
}
__device__ __forceinline__ float wave32_sum(float v) {
  v += __shfl_xor(v, 16, 32);
  return half16_sum(v);
}

// --- 1) L2-normalize rows, emit f16 reps + invnorm. One wave32 per row. -----
__global__ __launch_bounds__(256)
void ntx_normalize(const float* __restrict__ zi, const float* __restrict__ zj,
                   _Float16* __restrict__ reps, float* __restrict__ invnorm) {
  const int wave = threadIdx.x >> 5;
  const int lane = threadIdx.x & 31;
  const int row  = blockIdx.x * 8 + wave;                 // 1024 blocks * 8 rows
  const float* src = (row < NROWS) ? (zi + (size_t)row * DIM)
                                   : (zj + (size_t)(row - NROWS) * DIM);
  const float4* s4 = (const float4*)(src + lane * 8);
  float4 a = s4[0], b = s4[1];
  float ss = a.x*a.x + a.y*a.y + a.z*a.z + a.w*a.w
           + b.x*b.x + b.y*b.y + b.z*b.z + b.w*b.w;
  ss = wave32_sum(ss);
  const float inv = 1.0f / fmaxf(sqrtf(ss), 1e-12f);
  v8h o;
  o[0] = (_Float16)(a.x*inv); o[1] = (_Float16)(a.y*inv);
  o[2] = (_Float16)(a.z*inv); o[3] = (_Float16)(a.w*inv);
  o[4] = (_Float16)(b.x*inv); o[5] = (_Float16)(b.y*inv);
  o[6] = (_Float16)(b.z*inv); o[7] = (_Float16)(b.w*inv);
  *(v8h*)(reps + (size_t)row * DIM + lane * 8) = o;
  if (lane == 0) invnorm[row] = inv;
}

// --- 2) exact f32 positives. One wave32 per pair. ---------------------------
__global__ __launch_bounds__(256)
void ntx_positives(const float* __restrict__ zi, const float* __restrict__ zj,
                   const float* __restrict__ invnorm, float* __restrict__ pos) {
  const int wave = threadIdx.x >> 5;
  const int lane = threadIdx.x & 31;
  const int row  = blockIdx.x * 8 + wave;                 // 512 blocks * 8 rows
  const float4* a4 = (const float4*)(zi + (size_t)row * DIM + lane * 8);
  const float4* b4 = (const float4*)(zj + (size_t)row * DIM + lane * 8);
  float4 a0 = a4[0], a1 = a4[1], b0 = b4[0], b1 = b4[1];
  float d = a0.x*b0.x + a0.y*b0.y + a0.z*b0.z + a0.w*b0.w
          + a1.x*b1.x + a1.y*b1.y + a1.z*b1.z + a1.w*b1.w;
  d = wave32_sum(d);
  if (lane == 0) pos[row] = d * invnorm[row] * invnorm[row + NROWS];
}

// --- 3) fused  rowsum_r += sum_c exp(2 * (R @ R^T)[r,c])  -------------------
// Block: 8 waves = 128 rows x 1024 cols (grid 64 x 8). Per 32-col stage the
// block stages a 32x256 f16 B tile in LDS (double-buffered); each wave runs
// 16 WMMAs with a depth-2 fragment prefetch queue and alternating acc chains.
__global__ __launch_bounds__(256)
__attribute__((amdgpu_waves_per_eu(2)))
void ntx_simsum(const _Float16* __restrict__ reps, float* __restrict__ rowsum) {
  __shared__ _Float16 smem[2][32 * ROW_H];          // 2 x 16896 B

  const int tid  = threadIdx.x;
  const int wave = tid >> 5;
  const int lane = tid & 31;
  const int l15  = lane & 15;
  const int hs   = lane >> 4;                       // half-wave select
  const int m0   = (blockIdx.x * 8 + wave) * 16;    // this wave's 16 rows
  const int c0   = blockIdx.y * 1024;               // this block's 1024 columns

  // Preload A fragments for all 8 K-blocks (K = 256) -> held in 64 VGPRs.
  // A 16x32 f16 layout: lanes<16 take K {k0..k0+7, k0+16..k0+23},
  //                     lanes>=16 take K {k0+8..k0+15, k0+24..k0+31}.
  Frag afrag[8];
  {
    const _Float16* arow = reps + (size_t)(m0 + l15) * DIM + hs * 8;
    #pragma unroll
    for (int kb = 0; kb < 8; ++kb) {
      const _Float16* p = arow + kb * 32;
      afrag[kb].h[0] = *(const v8h*)(p);
      afrag[kb].h[1] = *(const v8h*)(p + 16);
    }
  }

  // Cooperative staging: thread tid copies 32 halfs (64B) of the 32-row x
  // 256-half tile: row = tid/8, segment = (tid%8)*32.
  const int crow = tid >> 3;                        // 0..31
  const int cseg = (tid & 7) * 32;                  // half offset within row
  const _Float16* gsrc = reps + (size_t)(c0 + crow) * DIM + cseg;
  const int loff = crow * ROW_H + cseg;             // same LDS offset both bufs

  {                                                 // prologue: stage 0 -> buf 0
    v8h g0 = *(const v8h*)(gsrc);
    v8h g1 = *(const v8h*)(gsrc + 8);
    v8h g2 = *(const v8h*)(gsrc + 16);
    v8h g3 = *(const v8h*)(gsrc + 24);
    _Float16* ld = &smem[0][0] + loff;
    *(v8h*)(ld)      = g0;  *(v8h*)(ld + 8)  = g1;
    *(v8h*)(ld + 16) = g2;  *(v8h*)(ld + 24) = g3;
  }
  __syncthreads();

  float partial[8] = {0.f,0.f,0.f,0.f,0.f,0.f,0.f,0.f};

  // Step s = 0..15 maps to sub = s&1 (16-col subtile), kb = s>>1 (K-block):
  // consecutive steps hit alternating accumulators -> no WMMA->WMMA RAW.
  // Fragment LDS address for (sub, kb): sbase + sub*16*ROW_H + kb*32.
  #define FRAG_OFF(s) ((((s) & 1) * 16 * ROW_H) + (((s) >> 1) * 32))

  for (int t = 0; t < 32; ++t) {                    // 32 stages of 32 columns
    const int cur = t & 1;
    // Issue stage t+1's global loads first; they fly under this stage's math.
    v8h g0, g1, g2, g3;
    const bool have_next = (t + 1) < 32;
    if (have_next) {
      const _Float16* gn = gsrc + (size_t)(t + 1) * 32 * DIM;
      g0 = *(const v8h*)(gn);
      g1 = *(const v8h*)(gn + 8);
      g2 = *(const v8h*)(gn + 16);
      g3 = *(const v8h*)(gn + 24);
    }

    // B 32x16 f16 frag: lane = column N (= LDS row sub*16+l15); lanes<16 hold
    // K {k0..k0+7, k0+16..k0+23}, lanes>=16 K {k0+8..k0+15, k0+24..k0+31}.
    const _Float16* sbase = &smem[cur][0] + l15 * ROW_H + hs * 8;
    v8f acc0 = {}, acc1 = {};
    Frag f0, f1;                                    // depth-2 prefetch queue
    f0.h[0] = *(const v8h*)(sbase + FRAG_OFF(0));
    f0.h[1] = *(const v8h*)(sbase + FRAG_OFF(0) + 16);
    f1.h[0] = *(const v8h*)(sbase + FRAG_OFF(1));
    f1.h[1] = *(const v8h*)(sbase + FRAG_OFF(1) + 16);
    #pragma unroll
    for (int s = 0; s < 16; ++s) {
      Frag fn;
      if (s < 14) {                                 // prefetch step s+2
        const _Float16* p = sbase + FRAG_OFF(s + 2);
        fn.h[0] = *(const v8h*)(p);
        fn.h[1] = *(const v8h*)(p + 16);
      }
      if ((s & 1) == 0)
        acc0 = __builtin_amdgcn_wmma_f32_16x16x32_f16(
            false, afrag[s >> 1].v, false, f0.v, (short)0, acc0, false, false);
      else
        acc1 = __builtin_amdgcn_wmma_f32_16x16x32_f16(
            false, afrag[s >> 1].v, false, f0.v, (short)0, acc1, false, false);
      f0 = f1;
      if (s < 14) f1 = fn;
    }

    // D layout: VGPR p, lanes<16 -> row m0+p, lanes>=16 -> row m0+8+p.
    #pragma unroll
    for (int p = 0; p < 8; ++p)
      partial[p] += __builtin_amdgcn_exp2f(acc0[p] * LOG2E_X2)
                  + __builtin_amdgcn_exp2f(acc1[p] * LOG2E_X2);

    if (have_next) {                                // publish stage t+1
      _Float16* ln = &smem[cur ^ 1][0] + loff;
      *(v8h*)(ln)      = g0;  *(v8h*)(ln + 8)  = g1;
      *(v8h*)(ln + 16) = g2;  *(v8h*)(ln + 24) = g3;
      __syncthreads();                              // one barrier per 32 cols
    }
  }
  #undef FRAG_OFF

  #pragma unroll
  for (int p = 0; p < 8; ++p) {
    float s = half16_sum(partial[p]);               // sum 16 columns per row
    if (l15 == 0) atomicAdd(&rowsum[m0 + p + hs * 8], s);
  }
}

// --- 4) final scalar: mean_k( log(rowsum_k - e^2) - 2*pos_k ) ---------------
__global__ __launch_bounds__(256)
void ntx_loss(const float* __restrict__ rowsum, const float* __restrict__ pos,
              float* __restrict__ out) {
  __shared__ float sh[256];
  float acc = 0.f;
  for (int k = threadIdx.x; k < TWO_N; k += 256) {
    const float denom = rowsum[k] - E2;             // mask self-similarity (==1)
    acc += logf(denom) - TEMP_INV * pos[k & (NROWS - 1)];
  }
  sh[threadIdx.x] = acc;
  __syncthreads();
  for (int s = 128; s > 0; s >>= 1) {
    if (threadIdx.x < s) sh[threadIdx.x] += sh[threadIdx.x + s];
    __syncthreads();
  }
  if (threadIdx.x == 0) out[0] = sh[0] * (1.0f / (float)TWO_N);
}

extern "C" void kernel_launch(void* const* d_in, const int* in_sizes, int n_in,
                              void* d_out, int out_size, void* d_ws, size_t ws_size,
                              hipStream_t stream) {
  const float* zi = (const float*)d_in[0];
  const float* zj = (const float*)d_in[1];

  char* ws = (char*)d_ws;
  _Float16* reps    = (_Float16*)(ws);                              // 8192*256*2 = 4 MiB
  float*    invnorm = (float*)(ws + 4194304);                       // 32 KiB
  float*    pos     = (float*)(ws + 4194304 + 32768);               // 16 KiB
  float*    rowsum  = (float*)(ws + 4194304 + 32768 + 16384);       // 32 KiB

  hipMemsetAsync(rowsum, 0, TWO_N * sizeof(float), stream);

  ntx_normalize<<<dim3(TWO_N / 8), dim3(256), 0, stream>>>(zi, zj, reps, invnorm);
  ntx_positives<<<dim3(NROWS / 8), dim3(256), 0, stream>>>(zi, zj, invnorm, pos);
  ntx_simsum  <<<dim3(TWO_N / 128, 8), dim3(256), 0, stream>>>(reps, rowsum);
  ntx_loss    <<<dim3(1), dim3(256), 0, stream>>>(rowsum, pos, (float*)d_out);
}
